// SpanBasedDynamicConvolution_2319282340579
// MI455X (gfx1250) — matrix operand order repaired and checked by reading, not compiled, not run on hardware
//
#include <hip/hip_runtime.h>

#define Bb 4
#define Ss 2048
#define Dd 768
#define Hh 6
#define HD 64
#define AHS 384
#define KK 9
#define PADW 4
#define Mtot (Bb * Ss)           // 8192
#define WSLOT (AHS * Dd)         // 294912 elems per weight slot

typedef __attribute__((ext_vector_type(16))) __bf16 v16bf;
typedef __attribute__((ext_vector_type(8)))  float  v8f;
typedef __attribute__((ext_vector_type(4)))  int    v4i;

union FragU { v16bf v; unsigned int u[8]; };

// ---------------- async global->LDS (gfx1250 ASYNCcnt path) ----------------
#if defined(__has_builtin)
#if __has_builtin(__builtin_amdgcn_global_load_async_to_lds_b128)
#define HAVE_ASYNC_LDS 1
#endif
#endif
#ifndef HAVE_ASYNC_LDS
#define HAVE_ASYNC_LDS 0
#endif

typedef __attribute__((address_space(1))) v4i as1_v4i;
typedef __attribute__((address_space(3))) v4i as3_v4i;

__device__ inline void copy16_g2l(const void* g, void* l) {
#if HAVE_ASYNC_LDS
    __builtin_amdgcn_global_load_async_to_lds_b128(
        (as1_v4i*)(uintptr_t)g,
        (as3_v4i*)(unsigned int)(uintptr_t)l, 0, 0);
#else
    *(int4*)l = *(const int4*)g;
#endif
}

__device__ inline void async_wait_all() {
#if HAVE_ASYNC_LDS
#if __has_builtin(__builtin_amdgcn_s_wait_asynccnt)
    __builtin_amdgcn_s_wait_asynccnt(0);
#else
    asm volatile("s_wait_asynccnt 0x0" ::: "memory");
#endif
#endif
}

__device__ inline unsigned short f2bf(float f) {
    union { float f; unsigned int u; } v; v.f = f;
    unsigned int r = v.u + 0x7FFFu + ((v.u >> 16) & 1u);
    return (unsigned short)(r >> 16);
}

// A-matrix 16x32 bf16 fragment from LDS stored row-major [m][k] with stride rs.
__device__ inline v16bf load_frag_A(const unsigned short* lds, int mBase, int rs,
                                    int kBase, int lane) {
    int m = mBase + (lane & 15);
    int kh = (lane >> 4) * 16;
    FragU f;
#pragma unroll
    for (int j = 0; j < 8; ++j) {
        int k = kBase + (j << 1) + kh;
        f.u[j] = *(const unsigned int*)&lds[m * rs + k];
    }
    return f.v;
}

// B-matrix 32x16 bf16 fragment: element (k,n) from LDS stored [n][k] stride ks.
__device__ inline v16bf load_frag_B(const unsigned short* lds, int nBase, int ks,
                                    int kBase, int lane) {
    int n = nBase + (lane & 15);
    int kh = (lane >> 4) * 16;
    FragU f;
#pragma unroll
    for (int j = 0; j < 8; ++j) {
        int k = kBase + (j << 1) + kh;
        f.u[j] = *(const unsigned int*)&lds[n * ks + k];
    }
    return f.v;
}

// ---------------- weight conversion: pack [Wq,Wk,Wv,Wco,pw] as bf16 --------
__global__ __launch_bounds__(256) void cvtw_kernel(
    const float* Wq, const float* Wk, const float* Wv, const float* Wco,
    const float* pw, unsigned short* wbf) {
    int which = blockIdx.y;
    const float* src = (which == 0) ? Wq : (which == 1) ? Wk :
                       (which == 2) ? Wv : (which == 3) ? Wco : pw;
    int i = blockIdx.x * 256 + threadIdx.x;
    if (i < WSLOT) wbf[(size_t)which * WSLOT + i] = f2bf(src[i]);
}

// ---------------- depthwise conv over S + bf16 conversion of x -------------
__global__ __launch_bounds__(256) void dwconv_kernel(
    const float* x, const float* dww, unsigned short* xbf, unsigned short* depbf) {
    size_t idx = (size_t)blockIdx.x * 256 + threadIdx.x;
    if (idx >= (size_t)Mtot * Dd) return;
    int d = (int)(idx % Dd);
    int s = (int)((idx / Dd) % Ss);
    float acc = 0.f;
#pragma unroll
    for (int k = 0; k < KK; ++k) {
        int ss = s + k - PADW;
        if (ss >= 0 && ss < Ss)
            acc += x[idx + (size_t)(k - PADW) * Dd] * dww[d * KK + k];
    }
    xbf[idx]   = f2bf(x[idx]);
    depbf[idx] = f2bf(acc);
}

// ---------------- WMMA projection GEMM: OUT[M,384] = A[M,768] x W^T --------
__global__ __launch_bounds__(256) void gemm_proj_kernel(
    const unsigned short* xbf, const unsigned short* depbf, const unsigned short* wbf,
    const float* bq, const float* bk, const float* bv, const float* bco,
    const float* sepb,
    float* qf, unsigned short* qh, unsigned short* kh, unsigned short* vh,
    float* cof, float* kcf) {
    const unsigned short* A; const float* bias;
    float* outf = nullptr; unsigned short* outh = nullptr;
    int job = blockIdx.z;
    const unsigned short* W = wbf + (size_t)job * WSLOT;
    switch (job) {
        case 0: A = xbf;   bias = bq;   outf = qf;  outh = qh; break;
        case 1: A = xbf;   bias = bk;   outh = kh;             break;
        case 2: A = xbf;   bias = bv;   outh = vh;             break;
        case 3: A = xbf;   bias = bco;  outf = cof;            break;
        default:A = depbf; bias = sepb; outf = kcf;            break;
    }
    __shared__ unsigned short lA[128 * 64];   // 16 KB
    __shared__ unsigned short lB[64 * 64];    // 8 KB

    int t    = threadIdx.x;
    int lane = t & 31;
    int wid  = t >> 5;          // 8 waves
    int m0   = blockIdx.x * 128;
    int n0   = blockIdx.y * 64;
    int mW   = (wid & 3) * 32;  // wave tile: 32x32
    int nW   = (wid >> 2) * 32;

    v8f acc00 = 0, acc01 = 0, acc10 = 0, acc11 = 0;

    for (int kk = 0; kk < Dd; kk += 64) {
        {   // stage A 128x64: 32 elems (64 B) per thread
            int r = t >> 1, half = t & 1;
            const unsigned short* src = A + (size_t)(m0 + r) * Dd + kk + half * 32;
            unsigned short* dst = &lA[r * 64 + half * 32];
#pragma unroll
            for (int c = 0; c < 4; ++c) copy16_g2l(src + c * 8, dst + c * 8);
        }
        {   // stage B 64x64 of W (n-major so k-pairs are contiguous)
            int n = t >> 2, q = t & 3;
            const unsigned short* src = W + (size_t)(n0 + n) * Dd + kk + q * 16;
            unsigned short* dst = &lB[n * 64 + q * 16];
#pragma unroll
            for (int c = 0; c < 2; ++c) copy16_g2l(src + c * 8, dst + c * 8);
        }
        async_wait_all();
        __syncthreads();

#pragma unroll
        for (int ks = 0; ks < 64; ks += 32) {
            v16bf a0 = load_frag_A(lA, mW,      64, ks, lane);
            v16bf a1 = load_frag_A(lA, mW + 16, 64, ks, lane);
            v16bf b0 = load_frag_B(lB, nW,      64, ks, lane);
            v16bf b1 = load_frag_B(lB, nW + 16, 64, ks, lane);
            acc00 = __builtin_amdgcn_wmma_f32_16x16x32_bf16(false, a0, false, b0, (short)0, acc00, false, false);
            acc01 = __builtin_amdgcn_wmma_f32_16x16x32_bf16(false, a0, false, b1, (short)0, acc01, false, false);
            acc10 = __builtin_amdgcn_wmma_f32_16x16x32_bf16(false, a1, false, b0, (short)0, acc10, false, false);
            acc11 = __builtin_amdgcn_wmma_f32_16x16x32_bf16(false, a1, false, b1, (short)0, acc11, false, false);
        }
        __syncthreads();
    }

    int half = lane >> 4;
    v8f accs[2][2] = {{acc00, acc01}, {acc10, acc11}};
#pragma unroll
    for (int i = 0; i < 2; ++i) {
#pragma unroll
        for (int j = 0; j < 2; ++j) {
            int n = n0 + nW + j * 16 + (lane & 15);
            float bv_ = bias[n];
#pragma unroll
            for (int r = 0; r < 8; ++r) {
                int m = m0 + mW + i * 16 + r + 8 * half;
                float val = accs[i][j][r] + bv_;
                size_t o = (size_t)m * AHS + n;
                if (outf) outf[o] = val;
                if (outh) outh[o] = f2bf(val);
            }
        }
    }
}

// ---------------- dynamic kernel gen + softmax + lightweight conv ----------
__global__ __launch_bounds__(64) void dynconv_kernel(
    const float* qf, const float* kcf, const float* cof,
    const float* Wck, const float* bck, float* out) {
    __shared__ float ca[AHS];
    __shared__ float kern[Hh * KK];
    int bs = blockIdx.x;
    int b = bs / Ss, s = bs % Ss;
    int t = threadIdx.x;

    for (int i = t; i < AHS; i += 64)
        ca[i] = qf[(size_t)bs * AHS + i] * kcf[(size_t)bs * AHS + i];
    __syncthreads();

    if (t < Hh * KK) {
        float acc = bck[t];
        for (int a = 0; a < AHS; ++a) acc += Wck[t * AHS + a] * ca[a];
        kern[t] = acc;
    }
    __syncthreads();

    if (t < Hh) {
        float mx = -3.0e38f;
#pragma unroll
        for (int k = 0; k < KK; ++k) mx = fmaxf(mx, kern[t * KK + k]);
        float sum = 0.f;
        float e[KK];
#pragma unroll
        for (int k = 0; k < KK; ++k) { e[k] = __expf(kern[t * KK + k] - mx); sum += e[k]; }
        float inv = 1.f / sum;
#pragma unroll
        for (int k = 0; k < KK; ++k) kern[t * KK + k] = e[k] * inv;
    }
    __syncthreads();

    for (int o = t; o < AHS; o += 64) {
        int h = o >> 6;
        float acc = 0.f;
#pragma unroll
        for (int k = 0; k < KK; ++k) {
            int ss = s + k - PADW;
            if (ss >= 0 && ss < Ss)
                acc += cof[((size_t)b * Ss + ss) * AHS + o] * kern[h * KK + k];
        }
        out[(size_t)bs * 768 + AHS + o] = acc;
    }
}

// ---------------- flash attention: one block = (b, h, 64 query rows) -------
__global__ __launch_bounds__(128) void attn_kernel(
    const unsigned short* qh, const unsigned short* khp, const unsigned short* vhp,
    const float* mask, float* out) {
    __shared__ unsigned short sQ[64 * 64];     // [row][hd]   8 KB
    __shared__ unsigned short sK[32 * 64];     // [key][hd]   4 KB
    __shared__ unsigned short sVrow[32 * 64];  // [key][hd]   4 KB
    __shared__ unsigned short sVT[64 * 32];    // [hd][key]   4 KB
    __shared__ unsigned short sP[4][16 * 32];  //             4 KB

    int t = threadIdx.x, lane = t & 31, wid = t >> 5, half = lane >> 4;
    int qtile = blockIdx.x, h = blockIdx.y, b = blockIdx.z;
    int qbase = qtile * 64;
    int mB = wid * 16;

    {   // stage Q 64x64 (async): 32 elems per thread
        int r = t >> 1, hf = t & 1;
        const unsigned short* src =
            qh + ((size_t)(b * Ss + qbase + r)) * AHS + h * HD + hf * 32;
        unsigned short* dst = &sQ[r * 64 + hf * 32];
#pragma unroll
        for (int c = 0; c < 4; ++c) copy16_g2l(src + c * 8, dst + c * 8);
    }

    float m_run[8], l_run[8];
    v8f accO[4];
#pragma unroll
    for (int r = 0; r < 8; ++r) { m_run[r] = -1e30f; l_run[r] = 0.f; }
#pragma unroll
    for (int i = 0; i < 4; ++i) accO[i] = 0;

    for (int kt = 0; kt < Ss / 32; ++kt) {
        int kbase = kt * 32;
        __syncthreads();   // previous iteration consumers done
        {   // stage K and V tiles [32][64], 16 elems per thread each
            int r = t >> 2, q = t & 3;
            const unsigned short* sk =
                khp + ((size_t)(b * Ss + kbase + r)) * AHS + h * HD + q * 16;
            const unsigned short* sv =
                vhp + ((size_t)(b * Ss + kbase + r)) * AHS + h * HD + q * 16;
            unsigned short* dk = &sK[r * 64 + q * 16];
            unsigned short* dv = &sVrow[r * 64 + q * 16];
#pragma unroll
            for (int c = 0; c < 2; ++c) {
                copy16_g2l(sk + c * 8, dk + c * 8);
                copy16_g2l(sv + c * 8, dv + c * 8);
            }
        }
        async_wait_all();
        __syncthreads();
        {   // LDS transpose V -> sVT[hd][key]
#pragma unroll
            for (int c = 0; c < 2; ++c) {
                int id = t * 2 + c;          // 0..255
                int i = id >> 3;             // key 0..31
                int dg = (id & 7) * 8;       // hd group
                unsigned short tmp[8];
                *(int4*)tmp = *(const int4*)&sVrow[i * 64 + dg];
#pragma unroll
                for (int j = 0; j < 8; ++j) sVT[(dg + j) * 32 + i] = tmp[j];
            }
        }
        __syncthreads();

        v8f s0 = 0, s1 = 0;
#pragma unroll
        for (int ks = 0; ks < 2; ++ks) {
            v16bf aQ = load_frag_A(sQ, mB, 64, ks * 32, lane);
            v16bf b0 = load_frag_B(sK, 0,  64, ks * 32, lane);
            v16bf b1 = load_frag_B(sK, 16, 64, ks * 32, lane);
            s0 = __builtin_amdgcn_wmma_f32_16x16x32_bf16(false, aQ, false, b0, (short)0, s0, false, false);
            s1 = __builtin_amdgcn_wmma_f32_16x16x32_bf16(false, aQ, false, b1, (short)0, s1, false, false);
        }
        float mk0 = mask[(size_t)b * Ss + kbase + (lane & 15)];
        float mk1 = mask[(size_t)b * Ss + kbase + 16 + (lane & 15)];

        float p0[8], p1[8], alpha[8];
#pragma unroll
        for (int r = 0; r < 8; ++r) {
            p0[r] = s0[r] * 0.125f + mk0;
            p1[r] = s1[r] * 0.125f + mk1;
        }
#pragma unroll
        for (int r = 0; r < 8; ++r) {
            float mx = fmaxf(p0[r], p1[r]);
            mx = fmaxf(mx, __shfl_xor(mx, 1));
            mx = fmaxf(mx, __shfl_xor(mx, 2));
            mx = fmaxf(mx, __shfl_xor(mx, 4));
            mx = fmaxf(mx, __shfl_xor(mx, 8));
            float m_new = fmaxf(m_run[r], mx);
            alpha[r] = __expf(m_run[r] - m_new);
            p0[r] = __expf(p0[r] - m_new);
            p1[r] = __expf(p1[r] - m_new);
            float rs = p0[r] + p1[r];
            rs += __shfl_xor(rs, 1);
            rs += __shfl_xor(rs, 2);
            rs += __shfl_xor(rs, 4);
            rs += __shfl_xor(rs, 8);
            l_run[r] = l_run[r] * alpha[r] + rs;
            m_run[r] = m_new;
        }
#pragma unroll
        for (int i = 0; i < 4; ++i)
#pragma unroll
            for (int r = 0; r < 8; ++r) accO[i][r] *= alpha[r];

        // stage P (wave-private) in A-fragment-friendly row-major [16][32]
#pragma unroll
        for (int r = 0; r < 8; ++r) {
            int m = r + 8 * half;
            int n = lane & 15;
            sP[wid][m * 32 + n]      = f2bf(p0[r]);
            sP[wid][m * 32 + 16 + n] = f2bf(p1[r]);
        }
        v16bf aP = load_frag_A(&sP[wid][0], 0, 32, 0, lane);
#pragma unroll
        for (int i = 0; i < 4; ++i) {
            v16bf bV = load_frag_B(sVT, i * 16, 32, 0, lane);
            accO[i] = __builtin_amdgcn_wmma_f32_16x16x32_bf16(false, aP, false, bV, (short)0, accO[i], false, false);
        }
    }

    // epilogue: ctx = O / l, first half of the 768-wide output
#pragma unroll
    for (int i = 0; i < 4; ++i) {
#pragma unroll
        for (int r = 0; r < 8; ++r) {
            int m = mB + r + 8 * half;
            int d = i * 16 + (lane & 15);
            out[((size_t)(b * Ss + qbase + m)) * 768 + h * HD + d] = accO[i][r] / l_run[r];
        }
    }
}

extern "C" void kernel_launch(void* const* d_in, const int* in_sizes, int n_in,
                              void* d_out, int out_size, void* d_ws, size_t ws_size,
                              hipStream_t stream) {
    const float* x    = (const float*)d_in[0];
    const float* mask = (const float*)d_in[1];
    const float* Wq   = (const float*)d_in[2];
    const float* bq   = (const float*)d_in[3];
    const float* Wk   = (const float*)d_in[4];
    const float* bk   = (const float*)d_in[5];
    const float* Wv   = (const float*)d_in[6];
    const float* bv   = (const float*)d_in[7];
    const float* dww  = (const float*)d_in[8];
    const float* pw   = (const float*)d_in[9];
    const float* sepb = (const float*)d_in[10];
    const float* Wck  = (const float*)d_in[11];
    const float* bck  = (const float*)d_in[12];
    const float* Wco  = (const float*)d_in[13];
    const float* bco  = (const float*)d_in[14];
    float* out = (float*)d_out;

    char* ws = (char*)d_ws;
    size_t off = 0;
    auto take = [&](size_t bytes) {
        char* p = ws + off;
        off = (off + bytes + 255) & ~(size_t)255;
        return p;
    };
    unsigned short* xbf   = (unsigned short*)take((size_t)Mtot * Dd * 2);
    unsigned short* depbf = (unsigned short*)take((size_t)Mtot * Dd * 2);
    unsigned short* wbf   = (unsigned short*)take((size_t)5 * WSLOT * 2);
    float*          qf    = (float*)take((size_t)Mtot * AHS * 4);
    unsigned short* qhb   = (unsigned short*)take((size_t)Mtot * AHS * 2);
    unsigned short* khb   = (unsigned short*)take((size_t)Mtot * AHS * 2);
    unsigned short* vhb   = (unsigned short*)take((size_t)Mtot * AHS * 2);
    float*          cof   = (float*)take((size_t)Mtot * AHS * 4);
    float*          kcf   = (float*)take((size_t)Mtot * AHS * 4);

    cvtw_kernel<<<dim3((WSLOT + 255) / 256, 5), 256, 0, stream>>>(Wq, Wk, Wv, Wco, pw, wbf);
    dwconv_kernel<<<(Mtot * Dd + 255) / 256, 256, 0, stream>>>(x, dww, xbf, depbf);
    gemm_proj_kernel<<<dim3(Mtot / 128, AHS / 64, 5), 256, 0, stream>>>(
        xbf, depbf, wbf, bq, bk, bv, bco, sepb, qf, qhb, khb, vhb, cof, kcf);
    dynconv_kernel<<<Mtot, 64, 0, stream>>>(qf, kcf, cof, Wck, bck, out);
    attn_kernel<<<dim3(Ss / 64, Hh, Bb), 128, 0, stream>>>(qhb, khb, vhb, mask, out);
}
